// MixtureOfExperts_66228395705075
// MI455X (gfx1250) — compile-verified, matrix operand order checked by reading, not run on hardware
//
#include <hip/hip_runtime.h>
#include <math.h>

#define T_TOK 8192
#define DDIM  1024
#define FDIM  4096
#define NEXP  8

// ---- LDS strides (f16 elements); every row start stays 16B aligned
#define XT_STRIDE  1032   // 32 rows    (X tile, f16, K = D)
#define W2T_STRIDE 72     // 1024 rows  (W2 chunk transposed: [n][f])
#define W1T_STRIDE 136    // 2x64 rows  (W1 stage transposed:  [f][dd], double buffered)
#define HS_STRIDE  72     // 32 rows    (gelu(h) chunk)

#define SMEM_BYTES (2*(32*XT_STRIDE + 1024*W2T_STRIDE + 2*64*W1T_STRIDE + 32*HS_STRIDE) + 32*4 + 32*4)
#define MAX_TILES  ((2*T_TOK + 31)/32 + NEXP)

typedef __attribute__((ext_vector_type(8)))  _Float16 v8h;
typedef __attribute__((ext_vector_type(16))) _Float16 v16h;
typedef __attribute__((ext_vector_type(8)))  float    v8f;
typedef __attribute__((ext_vector_type(4)))  float    float4v;

__device__ __forceinline__ float gelu_f(float x) {
  return 0.5f * x * (1.0f + erff(x * 0.70710678118654752f));
}

// Load one 16x32 f16 WMMA fragment from a row-major f16 LDS matrix.
// A-matrix layout (ISA 7.12.2): lanes 0-15 hold row M=lane, K in {0..7}u{16..23};
// lanes 16-31 hold the same rows with K in {8..15}u{24..31}. B mirrors with N on lanes.
__device__ __forceinline__ v16h load_frag(const _Float16* base, int stride,
                                          int row0, int kbase) {
  const int lane = threadIdx.x & 31;
  const _Float16* p = base + (row0 + (lane & 15)) * stride + kbase + ((lane >> 4) << 3);
  union { v8h h[2]; v16h v; } u;
  u.h[0] = *(const v8h*)(p);
  u.h[1] = *(const v8h*)(p + 16);
  return u.v;
}

// ---------------- Kernel 1: gating (one wave per token) ----------------
__global__ __launch_bounds__(256) void moe_gate(const float* __restrict__ X,
                                                const float* __restrict__ Wg,
                                                const float* __restrict__ bg,
                                                int* __restrict__ ctrl,
                                                int* __restrict__ topIdx,
                                                float* __restrict__ topW) {
  const int wave = threadIdx.x >> 5;
  const int lane = threadIdx.x & 31;
  const int t = blockIdx.x * 8 + wave;
  if (t >= T_TOK) return;

  float acc[NEXP];
#pragma unroll
  for (int e = 0; e < NEXP; e++) acc[e] = 0.f;

  const float* xr = X + (size_t)t * DDIM;
  for (int d = lane; d < DDIM; d += 32) {
    float x = xr[d];
#pragma unroll
    for (int e = 0; e < NEXP; e++) acc[e] += x * Wg[e * DDIM + d];
  }
#pragma unroll
  for (int e = 0; e < NEXP; e++) {
#pragma unroll
    for (int o = 16; o > 0; o >>= 1) acc[e] += __shfl_xor(acc[e], o, 32);
  }
#pragma unroll
  for (int e = 0; e < NEXP; e++) acc[e] += bg[e];

  // top-2 (softmax + renormalize == pairwise sigmoid of the logit gap)
  int e1 = 0; float l1 = acc[0];
#pragma unroll
  for (int e = 1; e < NEXP; e++) { if (acc[e] > l1) { l1 = acc[e]; e1 = e; } }
  int e2 = -1; float l2 = -3.4e38f;
#pragma unroll
  for (int e = 0; e < NEXP; e++) { if (e != e1 && acc[e] > l2) { l2 = acc[e]; e2 = e; } }
  float p2 = __expf(l2 - l1);
  float s  = 1.f + p2;
  float w1 = 1.f / s, w2 = p2 / s;

  if (lane == 0) {
    topIdx[t] = e1 | (e2 << 8);
    topW[2 * t + 0] = w1;
    topW[2 * t + 1] = w2;
    atomicAdd(&ctrl[e1], 1);
    atomicAdd(&ctrl[e2], 1);
  }
}

// ---------------- Kernel 2: offsets + tile table (E=8, trivial) ----------------
__global__ void moe_scan(int* __restrict__ ctrl) {
  if (threadIdx.x == 0) {
    int o = 0, ts = 0;
    ctrl[8] = 0; ctrl[25] = 0;
    for (int e = 0; e < NEXP; e++) {
      o  += ctrl[e];             ctrl[8 + e + 1]  = o;
      ts += (ctrl[e] + 31) >> 5; ctrl[25 + e + 1] = ts;
    }
    ctrl[34] = ts;              // total tiles
  }
}

// ---------------- Kernel 3: scatter tokens into per-expert lists ----------------
__global__ __launch_bounds__(256) void moe_scatter(int* __restrict__ ctrl,
                                                   const int* __restrict__ topIdx,
                                                   const float* __restrict__ topW,
                                                   int* __restrict__ pairTok,
                                                   float* __restrict__ pairW) {
  const int t = blockIdx.x * 256 + threadIdx.x;
  if (t >= T_TOK) return;
  int packed = topIdx[t];
  int e0 = packed & 0xFF, e1 = (packed >> 8) & 0xFF;
  float w0 = topW[2 * t + 0], w1 = topW[2 * t + 1];
  int p0 = atomicAdd(&ctrl[17 + e0], 1);
  int i0 = ctrl[8 + e0] + p0;
  pairTok[i0] = t; pairW[i0] = w0;
  int p1 = atomicAdd(&ctrl[17 + e1], 1);
  int i1 = ctrl[8 + e1] + p1;
  pairTok[i1] = t; pairW[i1] = w1;
}

// ---------------- Kernel 4: fused routed expert MLP (f16 WMMA) ----------------
__global__ __launch_bounds__(256) void moe_gemm(const float* __restrict__ X,
                                                const float* __restrict__ W1,
                                                const float* __restrict__ b1,
                                                const float* __restrict__ W2,
                                                const float* __restrict__ b2,
                                                const int* __restrict__ ctrl,
                                                const int* __restrict__ pairTok,
                                                const float* __restrict__ pairW,
                                                float* __restrict__ out) {
  extern __shared__ char smem[];
  _Float16* Xt  = (_Float16*)smem;              // [32][XT_STRIDE]
  _Float16* W2t = Xt  + 32 * XT_STRIDE;         // [1024][W2T_STRIDE] (transposed [n][f])
  _Float16* W1t = W2t + 1024 * W2T_STRIDE;      // [2][64][W1T_STRIDE] (transposed [f][dd])
  _Float16* Hs  = W1t + 2 * 64 * W1T_STRIDE;    // [32][HS_STRIDE]
  int*      rTok = (int*)(Hs + 32 * HS_STRIDE);
  float*    rW   = (float*)(rTok + 32);

  const int total = ctrl[34];
  const int blk = blockIdx.x;
  if (blk >= total) return;

  int e = 0;
  while (e < NEXP - 1 && blk >= ctrl[25 + e + 1]) e++;
  const int ti   = blk - ctrl[25 + e];
  const int cnt  = ctrl[e];
  const int off  = ctrl[8 + e];
  const int base = ti * 32;

  const int tid = threadIdx.x;
  if (tid < 32) {
    int j = base + tid;
    if (j < cnt) { rTok[tid] = pairTok[off + j]; rW[tid] = pairW[off + j]; }
    else         { rTok[tid] = -1;               rW[tid] = 0.f; }
  }
  __syncthreads();

  // Gather + convert X tile to f16 in LDS.
  for (int i4 = tid * 4; i4 < 32 * DDIM; i4 += 256 * 4) {
    int m  = i4 >> 10;
    int d0 = i4 & (DDIM - 1);
    int tok = rTok[m];
    float4v x = {0.f, 0.f, 0.f, 0.f};
    if (tok >= 0) x = *(const float4v*)(X + (size_t)tok * DDIM + d0);
    _Float16* dst = Xt + m * XT_STRIDE + d0;
    dst[0] = (_Float16)x.x; dst[1] = (_Float16)x.y;
    dst[2] = (_Float16)x.z; dst[3] = (_Float16)x.w;
  }

  const int wave   = tid >> 5;
  const int lane   = tid & 31;
  const int m_blk  = (wave & 1) * 16;      // M block (shared by both GEMMs)
  const int f_blk  = (wave >> 1) * 16;     // GEMM-1 F block
  const int n_base = (wave >> 1) * 256;    // GEMM-2: 16 N-blocks of 16

  const float* W1e = W1 + (size_t)e * DDIM * FDIM;
  const float* W2e = W2 + (size_t)e * FDIM * DDIM;

  v8f acc2[16];
#pragma unroll
  for (int j = 0; j < 16; j++) acc2[j] = (v8f){0, 0, 0, 0, 0, 0, 0, 0};

  for (int fc = 0; fc < FDIM; fc += 64) {
    __syncthreads();    // prior chunk's GEMM-2 reads of W2t/Hs are done

    // Prefetch a slice of the next W2 chunk into L2 while we work on this one.
    if (fc + 64 < FDIM) {
      const char* pf = (const char*)(W2e + (size_t)(fc + 64 + (tid >> 2)) * DDIM)
                       + (size_t)(tid & 3) * 1024;
      __builtin_prefetch(pf, 0, 1);
    }

    // Stage W2[fc:fc+64, :] transposed -> W2t[n][f] (f16)
    for (int i4 = tid * 4; i4 < 64 * DDIM; i4 += 256 * 4) {
      int f  = i4 >> 10;
      int n0 = i4 & (DDIM - 1);
      float4v w = *(const float4v*)(W2e + (size_t)(fc + f) * DDIM + n0);
      W2t[(n0 + 0) * W2T_STRIDE + f] = (_Float16)w.x;
      W2t[(n0 + 1) * W2T_STRIDE + f] = (_Float16)w.y;
      W2t[(n0 + 2) * W2T_STRIDE + f] = (_Float16)w.z;
      W2t[(n0 + 3) * W2T_STRIDE + f] = (_Float16)w.w;
    }

    // Stage W1 K-group 0 into buffer 0.
    {
      _Float16* buf = W1t;
#pragma unroll
      for (int q = 0; q < 8; q++) {
        int i4 = tid * 4 + q * 1024;
        int dd = i4 >> 6;
        int f0 = i4 & 63;
        float4v w = *(const float4v*)(W1e + (size_t)dd * FDIM + fc + f0);
        buf[(f0 + 0) * W1T_STRIDE + dd] = (_Float16)w.x;
        buf[(f0 + 1) * W1T_STRIDE + dd] = (_Float16)w.y;
        buf[(f0 + 2) * W1T_STRIDE + dd] = (_Float16)w.z;
        buf[(f0 + 3) * W1T_STRIDE + dd] = (_Float16)w.w;
      }
    }
    __syncthreads();

    // GEMM-1: h[32,64] = X[32,1024] @ W1[:, fc:fc+64]
    // Double-buffered K-groups: issue next group's global loads before the
    // WMMAs, convert/store after them (overlaps HBM latency with matrix math).
    v8f acc1 = (v8f){0, 0, 0, 0, 0, 0, 0, 0};
    for (int kg = 0; kg < 8; kg++) {
      const _Float16* cur = W1t + (kg & 1) * 64 * W1T_STRIDE;
      _Float16*       alt = W1t + ((kg & 1) ^ 1) * 64 * W1T_STRIDE;

      float4v staged[8];
      if (kg < 7) {
#pragma unroll
        for (int q = 0; q < 8; q++) {
          int i4 = tid * 4 + q * 1024;
          int dd = i4 >> 6;
          int f0 = i4 & 63;
          staged[q] = *(const float4v*)(W1e + (size_t)((kg + 1) * 128 + dd) * FDIM + fc + f0);
        }
      }

      // 4 WMMA k-steps, fragments pipelined one ahead.
      v16h a0 = load_frag(Xt, XT_STRIDE, m_blk, kg * 128);
      v16h b0 = load_frag(cur, W1T_STRIDE, f_blk, 0);
#pragma unroll
      for (int ks = 0; ks < 4; ks++) {
        v16h a1 = a0, b1v = b0;
        if (ks < 3) {
          a1  = load_frag(Xt, XT_STRIDE, m_blk, kg * 128 + (ks + 1) * 32);
          b1v = load_frag(cur, W1T_STRIDE, f_blk, (ks + 1) * 32);
        }
        acc1 = __builtin_amdgcn_wmma_f32_16x16x32_f16(false, a0, false, b0,
                                                      (short)0, acc1, false, false);
        a0 = a1; b0 = b1v;
      }

      if (kg < 7) {
#pragma unroll
        for (int q = 0; q < 8; q++) {
          int i4 = tid * 4 + q * 1024;
          int dd = i4 >> 6;
          int f0 = i4 & 63;
          alt[(f0 + 0) * W1T_STRIDE + dd] = (_Float16)staged[q].x;
          alt[(f0 + 1) * W1T_STRIDE + dd] = (_Float16)staged[q].y;
          alt[(f0 + 2) * W1T_STRIDE + dd] = (_Float16)staged[q].z;
          alt[(f0 + 3) * W1T_STRIDE + dd] = (_Float16)staged[q].w;
        }
      }
      __syncthreads();
    }

    // bias + exact GELU -> Hs (f16)
    {
      int   ncol = f_blk + (lane & 15);
      float bias = b1[(size_t)e * FDIM + fc + ncol];
      int   mrow = m_blk + ((lane >> 4) << 3);
#pragma unroll
      for (int r = 0; r < 8; r++) {
        float h = acc1[r] + bias;
        Hs[(mrow + r) * HS_STRIDE + ncol] = (_Float16)gelu_f(h);
      }
    }
    __syncthreads();

    // GEMM-2: out[32,1024] += gelu(h)[32,64] @ W2[fc:fc+64, :]
#pragma unroll
    for (int s = 0; s < 2; s++) {
      v16h a2   = load_frag(Hs, HS_STRIDE, m_blk, s * 32);
      v16h bcur = load_frag(W2t, W2T_STRIDE, n_base, s * 32);
#pragma unroll
      for (int j = 0; j < 16; j++) {
        v16h bnxt = bcur;
        if (j < 15) bnxt = load_frag(W2t, W2T_STRIDE, n_base + (j + 1) * 16, s * 32);
        acc2[j] = __builtin_amdgcn_wmma_f32_16x16x32_f16(false, a2, false, bcur,
                                                         (short)0, acc2[j], false, false);
        bcur = bnxt;
      }
    }
  }

  // Epilogue: (+b2) * routing weight, atomic accumulate into out.
  const int mrow0 = m_blk + ((lane >> 4) << 3);
#pragma unroll
  for (int j = 0; j < 16; j++) {
    int   n     = n_base + j * 16 + (lane & 15);
    float bias2 = b2[(size_t)e * DDIM + n];
#pragma unroll
    for (int r = 0; r < 8; r++) {
      int m = mrow0 + r;
      int tok = rTok[m];
      if (tok >= 0) {
        float v = (acc2[j][r] + bias2) * rW[m];
        __hip_atomic_fetch_add(out + (size_t)tok * DDIM + n, v,
                               __ATOMIC_RELAXED, __HIP_MEMORY_SCOPE_AGENT);
      }
    }
  }
}

extern "C" void kernel_launch(void* const* d_in, const int* in_sizes, int n_in,
                              void* d_out, int out_size, void* d_ws, size_t ws_size,
                              hipStream_t stream) {
  const float* X  = (const float*)d_in[0];
  const float* Wg = (const float*)d_in[1];
  const float* bg = (const float*)d_in[2];
  const float* W1 = (const float*)d_in[3];
  const float* b1 = (const float*)d_in[4];
  const float* W2 = (const float*)d_in[5];
  const float* b2 = (const float*)d_in[6];
  float* out = (float*)d_out;

  char*  ws      = (char*)d_ws;
  int*   ctrl    = (int*)ws;                               // 64 ints (counts/offsets/cursors/tiles)
  int*   pairTok = (int*)(ws + 256);                       // 2T ints
  float* pairW   = (float*)(ws + 256 + 2 * T_TOK * 4);     // 2T floats
  int*   topIdx  = (int*)(ws + 256 + 4 * T_TOK * 4);       // T ints
  float* topW    = (float*)(ws + 256 + 5 * T_TOK * 4);     // 2T floats

  hipMemsetAsync(d_ws, 0, 256, stream);                          // counters/cursors
  hipMemsetAsync(d_out, 0, (size_t)out_size * sizeof(float), stream);

  moe_gate   <<<T_TOK / 8,   256, 0, stream>>>(X, Wg, bg, ctrl, topIdx, topW);
  moe_scan   <<<1,           32,  0, stream>>>(ctrl);
  moe_scatter<<<T_TOK / 256, 256, 0, stream>>>(ctrl, topIdx, topW, pairTok, pairW);
  moe_gemm   <<<MAX_TILES,   256, SMEM_BYTES, stream>>>(X, W1, b1, W2, b2,
                                                        ctrl, pairTok, pairW, out);
}
